// LigerLMHeadJSD_37958920962547
// MI455X (gfx1250) — matrix operand order repaired and checked by reading, not compiled, not run on hardware
//
#include <hip/hip_runtime.h>

// ---------------------------------------------------------------------------
// Fused linear + beta-skew JSD (Liger fused_linear_jsd), MI455X / gfx1250.
//   s_logits = student_input @ student_weight^T   [2048 x 32000], K=2048
//   t_logits = teacher_input @ teacher_weight^T   [2048 x 32000], K=4096
//   loss = mean over non-ignored tokens of sum_v 0.5*P*(Y-logM)+0.5*Q*(X-logM)
// GEMMs: bf16 WMMA (v_wmma_f32_16x16x32_bf16) with f32 accumulation,
// register-staged + double-buffered LDS software pipeline.
// ---------------------------------------------------------------------------

#define BT_DIM 2048
#define H_DIM  4096
#define V_DIM  32000
#define BETA   0.5f
#define IGNORE_INDEX (-100)

typedef __attribute__((ext_vector_type(16))) __bf16 v16bf;
typedef __attribute__((ext_vector_type(8)))  __bf16 v8bf;
typedef __attribute__((ext_vector_type(4)))  __bf16 v4bf;
typedef __attribute__((ext_vector_type(4)))  float  v4f;
typedef __attribute__((ext_vector_type(8)))  float  v8f;

#define BM 128
#define BN 128
#define BK 32

// Native packed f32 -> bf16 conversion: one vector fptrunc lowers to two
// v_cvt_pk_bf16_f32 (RNE), each already producing a packed dword.
__device__ __forceinline__ ushort4 pack_bf16x4(float4 v) {
  v4f f = __builtin_bit_cast(v4f, v);
  v4bf b = __builtin_convertvector(f, v4bf);
  return __builtin_bit_cast(ushort4, b);
}

// C[M x N] = A[M x K] * B[N x K]^T   (both A and B are K-contiguous, TN GEMM)
__global__ __launch_bounds__(256) void gemm_bf16_wmma_kernel(
    const float* __restrict__ A,   // [M x K]
    const float* __restrict__ B,   // [N x K] (weights, row-major over K)
    float* __restrict__ C,         // [M x N]
    int M, int N, int K)
{
  // Double-buffered LDS tiles in bf16; row stride 32 elems (64B-aligned rows).
  __shared__ __align__(64) unsigned short As[2][BM][BK];
  __shared__ __align__(64) unsigned short Bs[2][BN][BK];

  const int tid  = threadIdx.x;
  const int lane = tid & 31;
  const int wave = tid >> 5;      // 0..7
  const int wm   = wave >> 1;     // 0..3 -> M offset within tile
  const int wn   = wave & 1;      // 0..1 -> N offset within tile
  const int m0   = wm * 32;       // wave sub-tile: 32 (M) x 64 (N)
  const int n0   = wn * 64;

  const int blockN = blockIdx.x * BN;
  const int blockM = blockIdx.y * BM;

  const int lo16   = lane & 15;
  const int hiHalf = lane >> 4;   // 0 for lanes 0-15, 1 for lanes 16-31

  // Per-thread global tile slots: slot s -> row s>>3, col (s&7)*4
  const int ldRow = tid >> 3;           // base row (+32 per slot step)
  const int ldC4  = (tid & 7) << 2;     // col 0,4,...,28
  const float* aBase = A + (size_t)(blockM + ldRow) * K + ldC4;
  const float* bBase = B + (size_t)(blockN + ldRow) * K + ldC4;

  v8f acc[2][4] = {};
  float4 ra[4], rb[4];

  // ---- prologue: stage 0 -> regs -> LDS[0] ----
#pragma unroll
  for (int i = 0; i < 4; ++i) {
    ra[i] = *(const float4*)(aBase + (size_t)(i * 32) * K);
    rb[i] = *(const float4*)(bBase + (size_t)(i * 32) * K);
  }
#pragma unroll
  for (int i = 0; i < 4; ++i) {
    *(ushort4*)&As[0][ldRow + i * 32][ldC4] = pack_bf16x4(ra[i]);
    *(ushort4*)&Bs[0][ldRow + i * 32][ldC4] = pack_bf16x4(rb[i]);
  }
  __syncthreads();

  int buf = 0;
  for (int k0 = 0; k0 < K; k0 += BK) {
    const bool hasNext = (k0 + BK) < K;

    // ---- issue next-stage global loads (latency hidden behind WMMAs) ----
    if (hasNext) {
#pragma unroll
      for (int i = 0; i < 4; ++i) {
        ra[i] = *(const float4*)(aBase + (size_t)(i * 32) * K + (k0 + BK));
        rb[i] = *(const float4*)(bBase + (size_t)(i * 32) * K + (k0 + BK));
      }
      // distance-2 prefetch on the streamed weight panel
      if (k0 + 2 * BK < K)
        __builtin_prefetch(bBase + (k0 + 2 * BK), 0, 1);
    }

    // ---- compute stage from LDS[buf]: 8 WMMAs per wave ----
    // A fragment (16-bit A 16x32): lane L holds row m=L%16,
    //   K = 0..15 for lanes 0-15, 16..31 for lanes 16-31.
    union Frag { v16bf v; v8bf h[2]; };
    v16bf afrag[2];
#pragma unroll
    for (int fm = 0; fm < 2; ++fm) {
      int m  = m0 + fm * 16 + lo16;
      int kh = hiHalf << 4;          // 0 or 16
      Frag fa;
      fa.h[0] = *(const v8bf*)&As[buf][m][kh];        // K = kh .. kh+7
      fa.h[1] = *(const v8bf*)&As[buf][m][kh + 8];    // K = kh+8 .. kh+15
      afrag[fm] = fa.v;
    }
    // B fragment (16-bit B 32x16): lane L holds col n=L%16,
    //   K = 0..15 (lanes 0-15) or 16..31 (lanes 16-31): 16 contiguous bf16.
#pragma unroll
    for (int fn = 0; fn < 4; ++fn) {
      int n  = n0 + fn * 16 + lo16;
      int kb = hiHalf << 4;          // 0 or 16
      v16bf bfrag = *(const v16bf*)&Bs[buf][n][kb];
#pragma unroll
      for (int fm = 0; fm < 2; ++fm) {
        acc[fm][fn] = __builtin_amdgcn_wmma_f32_16x16x32_bf16(
            /*neg_a=*/false, afrag[fm], /*neg_b=*/false, bfrag,
            /*c_mod=*/(short)0, acc[fm][fn],
            /*reuse_a=*/false, /*reuse_b=*/false);
      }
    }

    // ---- convert staged regs -> LDS[buf^1] (waits on loads here) ----
    if (hasNext) {
      const int nb = buf ^ 1;
#pragma unroll
      for (int i = 0; i < 4; ++i) {
        *(ushort4*)&As[nb][ldRow + i * 32][ldC4] = pack_bf16x4(ra[i]);
        *(ushort4*)&Bs[nb][ldRow + i * 32][ldC4] = pack_bf16x4(rb[i]);
      }
      buf = nb;
    }
    __syncthreads();
  }

  // ---- epilogue: C/D layout = VGPR i: M=i (lanes 0-15), M=i+8 (lanes 16-31)
#pragma unroll
  for (int fm = 0; fm < 2; ++fm) {
#pragma unroll
    for (int fn = 0; fn < 4; ++fn) {
      int col   = blockN + n0 + fn * 16 + lo16;
      int rbase = blockM + m0 + fm * 16 + (hiHalf << 3);
#pragma unroll
      for (int i = 0; i < 8; ++i) {
        C[(size_t)(rbase + i) * N + col] = acc[fm][fn][i];
      }
    }
  }
}

// ---------------------------------------------------------------------------
// Per-row log-softmax (online LSE) + beta-skew JSD, one block per token row.
// ---------------------------------------------------------------------------
__device__ __forceinline__ void lse_combine(float& m, float& s, float m2, float s2) {
  float M = fmaxf(m, m2);
  s = s * __expf(m - M) + s2 * __expf(m2 - M);
  m = M;
}

__device__ __forceinline__ void lse_push(float& m, float& s, float x) {
  if (x > m) { s = s * __expf(m - x) + 1.f; m = x; }
  else       { s += __expf(x - m); }
}

__global__ __launch_bounds__(256) void jsd_row_kernel(
    const float* __restrict__ sLogits,
    const float* __restrict__ tLogits,
    const int*   __restrict__ target,
    float* __restrict__ tok_loss,
    int V)
{
  const int b    = blockIdx.x;
  const int tid  = threadIdx.x;
  const int lane = tid & 31;
  const int wave = tid >> 5;
  const float* srow = sLogits + (size_t)b * V;
  const float* trow = tLogits + (size_t)b * V;

  const int V4 = V & ~1023;   // vectorized span: 256 threads * 4 elems

  // ---- pass 1: online (max, sum-exp) for student and teacher ----
  float ms = -1e30f, ss = 0.f, mt = -1e30f, st = 0.f;
  for (int v = tid * 4; v < V4; v += 1024) {
    float4 xs = *(const float4*)(srow + v);
    float4 xt = *(const float4*)(trow + v);
    lse_push(ms, ss, xs.x); lse_push(ms, ss, xs.y);
    lse_push(ms, ss, xs.z); lse_push(ms, ss, xs.w);
    lse_push(mt, st, xt.x); lse_push(mt, st, xt.y);
    lse_push(mt, st, xt.z); lse_push(mt, st, xt.w);
  }
  for (int v = V4 + tid; v < V; v += 256) {
    lse_push(ms, ss, srow[v]);
    lse_push(mt, st, trow[v]);
  }
  // wave32 shuffle reduction
  for (int off = 16; off > 0; off >>= 1) {
    float m2 = __shfl_xor(ms, off, 32);
    float s2 = __shfl_xor(ss, off, 32);
    lse_combine(ms, ss, m2, s2);
    m2 = __shfl_xor(mt, off, 32);
    s2 = __shfl_xor(st, off, 32);
    lse_combine(mt, st, m2, s2);
  }
  __shared__ float red[4][8];
  __shared__ float lse[2];
  if (lane == 0) { red[0][wave] = ms; red[1][wave] = ss; red[2][wave] = mt; red[3][wave] = st; }
  __syncthreads();
  if (tid == 0) {
    float m = red[0][0], s = red[1][0];
    for (int w = 1; w < 8; ++w) lse_combine(m, s, red[0][w], red[1][w]);
    lse[0] = m + __logf(s);
    m = red[2][0]; s = red[3][0];
    for (int w = 1; w < 8; ++w) lse_combine(m, s, red[2][w], red[3][w]);
    lse[1] = m + __logf(s);
  }
  __syncthreads();
  const float s_lse = lse[0];
  const float t_lse = lse[1];

  // ---- pass 2: JSD term (row is L2-resident from pass 1) ----
  float acc = 0.f;
  auto jsd_term = [&](float xs, float xt) {
    float X = xs - s_lse;           // student log-prob
    float Y = xt - t_lse;           // teacher log-prob
    float Q = __expf(X);
    float P = __expf(Y);
    float Mx = BETA * P + (1.f - BETA) * Q;
    float logM = __logf(Mx);
    acc += BETA * P * (Y - logM) + (1.f - BETA) * Q * (X - logM);
  };
  for (int v = tid * 4; v < V4; v += 1024) {
    float4 xs = *(const float4*)(srow + v);
    float4 xt = *(const float4*)(trow + v);
    jsd_term(xs.x, xt.x); jsd_term(xs.y, xt.y);
    jsd_term(xs.z, xt.z); jsd_term(xs.w, xt.w);
  }
  for (int v = V4 + tid; v < V; v += 256) jsd_term(srow[v], trow[v]);

  for (int off = 16; off > 0; off >>= 1) acc += __shfl_xor(acc, off, 32);
  __shared__ float sred[8];
  if (lane == 0) sred[wave] = acc;
  __syncthreads();
  if (tid == 0) {
    float tot = 0.f;
    for (int w = 0; w < 8; ++w) tot += sred[w];
    tok_loss[b] = (target[b] != IGNORE_INDEX) ? tot : 0.f;
  }
}

// ---------------------------------------------------------------------------
// Final deterministic reduction: loss = sum(tok_loss) / max(#non-ignored, 1)
// ---------------------------------------------------------------------------
__global__ __launch_bounds__(256) void final_reduce_kernel(
    const float* __restrict__ tok_loss,
    const int*   __restrict__ target,
    float* __restrict__ out,
    int BT)
{
  const int tid  = threadIdx.x;
  const int lane = tid & 31;
  const int wave = tid >> 5;
  float sum = 0.f;
  int   cnt = 0;
  for (int i = tid; i < BT; i += 256) {
    sum += tok_loss[i];
    cnt += (target[i] != IGNORE_INDEX) ? 1 : 0;
  }
  for (int off = 16; off > 0; off >>= 1) {
    sum += __shfl_xor(sum, off, 32);
    cnt += __shfl_xor(cnt, off, 32);
  }
  __shared__ float ssum[8];
  __shared__ int   scnt[8];
  if (lane == 0) { ssum[wave] = sum; scnt[wave] = cnt; }
  __syncthreads();
  if (tid == 0) {
    float t = 0.f; int c = 0;
    for (int w = 0; w < 8; ++w) { t += ssum[w]; c += scnt[w]; }
    if (c < 1) c = 1;
    out[0] = t / (float)c;
  }
}

// ---------------------------------------------------------------------------
// Launch
// ---------------------------------------------------------------------------
extern "C" void kernel_launch(void* const* d_in, const int* in_sizes, int n_in,
                              void* d_out, int out_size, void* d_ws, size_t ws_size,
                              hipStream_t stream) {
  const float* student_input  = (const float*)d_in[0]; // [BT, H/2]
  const float* student_weight = (const float*)d_in[1]; // [V, H/2]
  const float* teacher_input  = (const float*)d_in[2]; // [BT, H]
  const float* teacher_weight = (const float*)d_in[3]; // [V, H]
  const int*   target         = (const int*)d_in[4];   // [BT]

  // workspace layout: s_logits | t_logits | tok_loss
  float* sLogits  = (float*)d_ws;
  float* tLogits  = sLogits + (size_t)BT_DIM * V_DIM;
  float* tok_loss = tLogits + (size_t)BT_DIM * V_DIM;

  dim3 block(256);
  dim3 grid(V_DIM / BN, BT_DIM / BM);   // 250 x 16 tiles

  // student: K = H/2 = 2048
  gemm_bf16_wmma_kernel<<<grid, block, 0, stream>>>(
      student_input, student_weight, sLogits, BT_DIM, V_DIM, H_DIM / 2);
  // teacher: K = H = 4096
  gemm_bf16_wmma_kernel<<<grid, block, 0, stream>>>(
      teacher_input, teacher_weight, tLogits, BT_DIM, V_DIM, H_DIM);

  jsd_row_kernel<<<BT_DIM, block, 0, stream>>>(sLogits, tLogits, target, tok_loss, V_DIM);

  final_reduce_kernel<<<1, block, 0, stream>>>(tok_loss, target, (float*)d_out, BT_DIM);
}
